// CustomDOMINANT_31997506355976
// MI455X (gfx1250) — compile-verified, matrix operand order checked by reading, not run on hardware
//
#include <hip/hip_runtime.h>

// ---------------------------------------------------------------------------
// DOMINANT graph autoencoder for MI455X (gfx1250, wave32, WMMA).
//   A_hat = sigmoid(hs hs^T) : 604 MB store-bound -> bf16 WMMA, NT stores
//   X_hat, z                 : GCN layers = bf16 WMMA GEMM + CSR SpMM
// ---------------------------------------------------------------------------

#define NN   12288
#define EE   393216
#define FIN  512
#define HH   64

typedef __attribute__((ext_vector_type(16))) __bf16       v16bf;
typedef __attribute__((ext_vector_type(8)))  float        v8f;
typedef __attribute__((ext_vector_type(4)))  unsigned int v4u;

union BFrag { v16bf bf; v4u q[2]; };
union AccU  { v8f v; float f[8]; };

__device__ __forceinline__ unsigned short f2bf(float f) {
    // round-to-nearest-even f32 -> bf16
    unsigned int u = __float_as_uint(f);
    unsigned int r = (u + 0x7FFFu + ((u >> 16) & 1u)) >> 16;
    return (unsigned short)r;
}

// ---------------------------------------------------------------------------
// row_ptr[i] = lower_bound(adj_rows, i)  (adj_rows is sorted)
// ---------------------------------------------------------------------------
__global__ void build_row_ptr(const int* __restrict__ rows, int e, int n,
                              int* __restrict__ row_ptr) {
    int i = blockIdx.x * blockDim.x + threadIdx.x;
    if (i > n) return;
    int lo = 0, hi = e;
    while (lo < hi) {
        int mid = (lo + hi) >> 1;
        if (rows[mid] < i) lo = mid + 1; else hi = mid;
    }
    row_ptr[i] = lo;
}

// ---------------------------------------------------------------------------
// elementwise f32 -> bf16
// ---------------------------------------------------------------------------
__global__ void cvt_f32_bf16(const float* __restrict__ in,
                             unsigned short* __restrict__ out, int n) {
    int i = blockIdx.x * 256 + threadIdx.x;
    if (i < n) out[i] = f2bf(in[i]);
}

// ---------------------------------------------------------------------------
// transpose + convert: in [K,Nc] f32 -> out [Nc,K] bf16
// ---------------------------------------------------------------------------
__global__ void cvtT_f32_bf16(const float* __restrict__ in,
                              unsigned short* __restrict__ out, int K, int Nc) {
    int i = blockIdx.x * 256 + threadIdx.x;
    if (i < K * Nc) {
        int k = i / Nc, n = i - k * Nc;
        out[(size_t)n * K + k] = f2bf(in[i]);
    }
}

__global__ void copy_f32_nt(const float* __restrict__ in, float* __restrict__ out, int n) {
    int i = blockIdx.x * 256 + threadIdx.x;
    if (i < n) __builtin_nontemporal_store(in[i], &out[i]);
}

// ---------------------------------------------------------------------------
// C[M,Nc] = A[M,K] * BT[Nc,K]^T   (both operands bf16 row-major, f32 accum)
// act: 0 = none (RT stores, C is re-read by SpMM),
//      1 = sigmoid + non-temporal stores (A_hat: 604 MB write-once stream).
// Each wave computes a 16x32 tile (one shared A fragment, two B fragments ->
// 2 WMMAs per 32-wide K chunk). 8 waves (256 thr) per block -> 128x32/block.
// Fragment loads follow the CDNA5 16-bit A(16x32)/B(32x16) VGPR layouts:
//   lane L (0-15) owns row/col L with K packs {k0..k0+7, k0+16..k0+23},
//   lane L+16 owns {k0+8..k0+15, k0+24..k0+31}  -> two 16B contiguous loads.
// ---------------------------------------------------------------------------
__global__ __launch_bounds__(256)
void gemm_bf16_tn(const unsigned short* __restrict__ A,
                  const unsigned short* __restrict__ BT,
                  float* __restrict__ C,
                  int M, int Nc, int K, int act) {
    const int lane   = threadIdx.x & 31;
    const int wid    = threadIdx.x >> 5;
    const int base_m = (blockIdx.y * 8 + wid) * 16;
    const int base_n = blockIdx.x * 32;
    const int half   = lane >> 4;   // K sub-block selector
    const int idx    = lane & 15;   // M (for A) / N (for B) index

    const unsigned short* arow  = A  + (size_t)(base_m + idx) * K;
    const unsigned short* brow0 = BT + (size_t)(base_n + idx) * K;
    const unsigned short* brow1 = BT + (size_t)(base_n + 16 + idx) * K;

    AccU acc0, acc1;
#pragma unroll
    for (int i = 0; i < 8; ++i) { acc0.f[i] = 0.0f; acc1.f[i] = 0.0f; }

    for (int k0 = 0; k0 < K; k0 += 32) {
        BFrag fa, fb0, fb1;
        const int o0 = k0 + half * 8;
        const int o1 = k0 + 16 + half * 8;
        fa.q[0]  = *(const v4u*)(arow  + o0);
        fa.q[1]  = *(const v4u*)(arow  + o1);
        fb0.q[0] = *(const v4u*)(brow0 + o0);
        fb0.q[1] = *(const v4u*)(brow0 + o1);
        fb1.q[0] = *(const v4u*)(brow1 + o0);
        fb1.q[1] = *(const v4u*)(brow1 + o1);
        acc0.v = __builtin_amdgcn_wmma_f32_16x16x32_bf16(
            false, fa.bf, false, fb0.bf, (short)0, acc0.v, false, false);
        acc1.v = __builtin_amdgcn_wmma_f32_16x16x32_bf16(
            false, fa.bf, false, fb1.bf, (short)0, acc1.v, false, false);
    }

    // C/D layout: VGPR r -> lanes 0-15: M=r, lanes 16-31: M=r+8; N = lane&15.
    // Per r, lanes cover 2x 128B contiguous row segments (cols base_n..+31).
#pragma unroll
    for (int r = 0; r < 8; ++r) {
        const size_t rowoff = (size_t)(base_m + r + half * 8) * Nc;
        float v0 = acc0.f[r];
        float v1 = acc1.f[r];
        if (act == 1) {
            v0 = 1.0f / (1.0f + __expf(-v0));
            v1 = 1.0f / (1.0f + __expf(-v1));
            __builtin_nontemporal_store(v0, &C[rowoff + base_n + idx]);
            __builtin_nontemporal_store(v1, &C[rowoff + base_n + 16 + idx]);
        } else {
            C[rowoff + base_n + idx]      = v0;
            C[rowoff + base_n + 16 + idx] = v1;
        }
    }
}

// ---------------------------------------------------------------------------
// CSR SpMM, 64 output features: one wave per row, 2 cols per lane.
// out[r,:] = sum_e vals[e]*sup[cols[e],:] + bias  (optional relu)
// ---------------------------------------------------------------------------
__global__ __launch_bounds__(256)
void spmm_bias_act_64(const float* __restrict__ sup,
                      const int* __restrict__ cols,
                      const float* __restrict__ vals,
                      const int* __restrict__ row_ptr,
                      const float* __restrict__ bias,
                      float* __restrict__ out, int relu) {
    const int lane = threadIdx.x & 31;
    const int row  = blockIdx.x * 8 + (threadIdx.x >> 5);
    const int e0 = row_ptr[row];
    const int e1 = row_ptr[row + 1];
    float a0 = 0.0f, a1 = 0.0f;
    for (int e = e0; e < e1; ++e) {
        const int   c = cols[e];
        const float v = vals[e];
        const float* s = sup + (size_t)c * 64;
        a0 += v * s[lane];
        a1 += v * s[lane + 32];
    }
    a0 += bias[lane];
    a1 += bias[lane + 32];
    if (relu) { a0 = fmaxf(a0, 0.0f); a1 = fmaxf(a1, 0.0f); }
    out[(size_t)row * 64 + lane]      = a0;
    out[(size_t)row * 64 + lane + 32] = a1;
}

// ---------------------------------------------------------------------------
// CSR SpMM, 512 output features: one 256-thread block per row, 2 cols/thread.
// Writes X_hat (write-once output) -> non-temporal stores.
// ---------------------------------------------------------------------------
__global__ __launch_bounds__(256)
void spmm_bias_512(const float* __restrict__ sup,
                   const int* __restrict__ cols,
                   const float* __restrict__ vals,
                   const int* __restrict__ row_ptr,
                   const float* __restrict__ bias,
                   float* __restrict__ out) {
    const int row = blockIdx.x;
    const int f   = threadIdx.x;
    const int e0 = row_ptr[row], e1 = row_ptr[row + 1];
    float a0 = 0.0f, a1 = 0.0f;
    for (int e = e0; e < e1; ++e) {
        const int   c = cols[e];
        const float v = vals[e];
        const float* s = sup + (size_t)c * 512;
        a0 += v * s[f];
        a1 += v * s[f + 256];
    }
    __builtin_nontemporal_store(a0 + bias[f],       &out[(size_t)row * 512 + f]);
    __builtin_nontemporal_store(a1 + bias[f + 256], &out[(size_t)row * 512 + f + 256]);
}

// ---------------------------------------------------------------------------
extern "C" void kernel_launch(void* const* d_in, const int* in_sizes, int n_in,
                              void* d_out, int out_size, void* d_ws, size_t ws_size,
                              hipStream_t stream) {
    const float* x        = (const float*)d_in[0];
    const int*   adj_rows = (const int*)  d_in[1];
    const int*   adj_cols = (const int*)  d_in[2];
    const float* adj_vals = (const float*)d_in[3];
    const float* w_e1 = (const float*)d_in[4];  const float* b_e1 = (const float*)d_in[5];
    const float* w_e2 = (const float*)d_in[6];  const float* b_e2 = (const float*)d_in[7];
    const float* w_a1 = (const float*)d_in[8];  const float* b_a1 = (const float*)d_in[9];
    const float* w_a2 = (const float*)d_in[10]; const float* b_a2 = (const float*)d_in[11];
    const float* w_s  = (const float*)d_in[12]; const float* b_s  = (const float*)d_in[13];

    float* out = (float*)d_out;
    float* A_hat = out;                                      // [N,N]
    float* X_hat = out + (size_t)NN * NN;                    // [N,512]
    float* z_out = out + (size_t)NN * NN + (size_t)NN * FIN; // [N,64]

    // ---- workspace carve-out (all 256B aligned) ----
    char* base = (char*)d_ws;
    size_t off = 0;
    auto carve = [&](size_t bytes) -> char* {
        char* p = base + off;
        off = (off + bytes + 255) & ~(size_t)255;
        return p;
    };
    int*            row_ptr = (int*)            carve((size_t)(NN + 1) * 4);
    unsigned short* a16     = (unsigned short*) carve((size_t)NN * FIN * 2);  // x16 / h16 / h2_16 / hs16
    unsigned short* z16     = (unsigned short*) carve((size_t)NN * HH * 2);
    unsigned short* wT16    = (unsigned short*) carve((size_t)FIN * HH * 2);
    float*          sup     = (float*)          carve((size_t)NN * FIN * 4);  // GEMM output, pre-SpMM
    float*          h       = (float*)          carve((size_t)NN * HH * 4);
    float*          z       = (float*)          carve((size_t)NN * HH * 4);
    float*          h2      = (float*)          carve((size_t)NN * HH * 4);
    float*          hs      = (float*)          carve((size_t)NN * HH * 4);
    (void)ws_size; (void)n_in; (void)in_sizes; (void)out_size;

    const int nW  = NN * HH;    // 786432
    const int nX  = NN * FIN;   // 6291456
    const int wKN = FIN * HH;   // 32768 (largest weight)

    // CSR row pointers
    build_row_ptr<<<(NN + 1 + 255) / 256, 256, 0, stream>>>(adj_rows, EE, NN, row_ptr);

    // ---- Layer e1: h = relu(SpMM(x @ w_e1) + b_e1) ----
    cvt_f32_bf16 <<<(nX + 255) / 256, 256, 0, stream>>>(x, a16, nX);
    cvtT_f32_bf16<<<(wKN + 255) / 256, 256, 0, stream>>>(w_e1, wT16, FIN, HH);
    gemm_bf16_tn <<<dim3(HH / 32, NN / 128), 256, 0, stream>>>(a16, wT16, sup, NN, HH, FIN, 0);
    spmm_bias_act_64<<<NN / 8, 256, 0, stream>>>(sup, adj_cols, adj_vals, row_ptr, b_e1, h, 1);

    // ---- Layer e2: z = SpMM(h @ w_e2) + b_e2 ----
    cvt_f32_bf16 <<<(nW + 255) / 256, 256, 0, stream>>>(h, a16, nW);
    cvtT_f32_bf16<<<(HH * HH + 255) / 256, 256, 0, stream>>>(w_e2, wT16, HH, HH);
    gemm_bf16_tn <<<dim3(HH / 32, NN / 128), 256, 0, stream>>>(a16, wT16, sup, NN, HH, HH, 0);
    spmm_bias_act_64<<<NN / 8, 256, 0, stream>>>(sup, adj_cols, adj_vals, row_ptr, b_e2, z, 0);

    cvt_f32_bf16<<<(nW + 255) / 256, 256, 0, stream>>>(z, z16, nW);

    // ---- Attribute decoder L1: h2 = relu(SpMM(z @ w_a1) + b_a1) ----
    cvtT_f32_bf16<<<(HH * HH + 255) / 256, 256, 0, stream>>>(w_a1, wT16, HH, HH);
    gemm_bf16_tn <<<dim3(HH / 32, NN / 128), 256, 0, stream>>>(z16, wT16, sup, NN, HH, HH, 0);
    spmm_bias_act_64<<<NN / 8, 256, 0, stream>>>(sup, adj_cols, adj_vals, row_ptr, b_a1, h2, 1);

    // ---- Attribute decoder L2: X_hat = SpMM(h2 @ w_a2) + b_a2 ----
    cvt_f32_bf16 <<<(nW + 255) / 256, 256, 0, stream>>>(h2, a16, nW);
    cvtT_f32_bf16<<<(wKN + 255) / 256, 256, 0, stream>>>(w_a2, wT16, HH, FIN);
    gemm_bf16_tn <<<dim3(FIN / 32, NN / 128), 256, 0, stream>>>(a16, wT16, sup, NN, FIN, HH, 0);
    spmm_bias_512<<<NN, 256, 0, stream>>>(sup, adj_cols, adj_vals, row_ptr, b_a2, X_hat);

    // ---- Structure decoder: hs = relu(SpMM(z @ w_s) + b_s) ----
    cvtT_f32_bf16<<<(HH * HH + 255) / 256, 256, 0, stream>>>(w_s, wT16, HH, HH);
    gemm_bf16_tn <<<dim3(HH / 32, NN / 128), 256, 0, stream>>>(z16, wT16, sup, NN, HH, HH, 0);
    spmm_bias_act_64<<<NN / 8, 256, 0, stream>>>(sup, adj_cols, adj_vals, row_ptr, b_s, hs, 1);

    // ---- A_hat = sigmoid(hs @ hs^T): the 604 MB store-bound GEMM ----
    cvt_f32_bf16<<<(nW + 255) / 256, 256, 0, stream>>>(hs, a16, nW);
    gemm_bf16_tn<<<dim3(NN / 32, NN / 128), 256, 0, stream>>>(a16, a16, A_hat, NN, NN, HH, 1);

    // ---- z passthrough output (write-once -> NT) ----
    copy_f32_nt<<<(nW + 255) / 256, 256, 0, stream>>>(z, z_out, nW);
}
